// GCN_26164940767949
// MI455X (gfx1250) — matrix operand order repaired and checked by reading, not compiled, not run on hardware
//
#include <hip/hip_runtime.h>

typedef __attribute__((ext_vector_type(16))) __bf16 v16bf;
typedef __attribute__((ext_vector_type(8)))  float  v8f;
typedef __attribute__((ext_vector_type(4)))  float  f4;

#define N_NODES 16384

// ---------------------------------------------------------------------------
// Feature transform + pack: Y[k][n] = sum_d H[k][d] * W[n][d], written
// directly in the v_wmma_f32_16x16x32_bf16 B-operand register image:
//   lane  = (n&15) + 16*((k>>3)&1)      (lanes 16..31 hold K=8..15,24..31)
//   elem  = (k&7) + 8*((k&31)>>4)       (16 bf16 per lane = 32B)
//   tile  = (k>>5)*NT + (n>>4)
// ---------------------------------------------------------------------------
template<int DIN, int DOUT>
__global__ __launch_bounds__(256)
void pack_feat(const float* __restrict__ H, const float* __restrict__ W,
               __bf16* __restrict__ Y) {
  const int idx = blockIdx.x * blockDim.x + threadIdx.x;   // over N*DOUT
  const int k = idx / DOUT;
  const int n = idx % DOUT;               // consecutive lanes share row k
  const float* __restrict__ h = H + (size_t)k * DIN;
  const float* __restrict__ w = W + (size_t)n * DIN;
  float s = 0.f;
#pragma unroll
  for (int d = 0; d < DIN; ++d) s += h[d] * w[d];

  constexpr int NT = DOUT / 16;
  const int ktile = k >> 5, kin = k & 31;
  const int ntile = n >> 4, nin = n & 15;
  const int lane  = nin + (((kin >> 3) & 1) << 4);
  const int e     = (kin & 7) + ((kin >> 4) << 3);
  Y[(((size_t)(ktile * NT + ntile) * 32 + lane) << 4) + e] = (__bf16)s;
}

// ---------------------------------------------------------------------------
// C[N,F] = (relu?) A[N,N] @ Ypacked[N,F]
// Block = 512 threads = 16 waves: 4 strips of 16 rows x 4-way K split
// (4096 waves total for HBM latency hiding). Each wave owns all F columns of
// its strip+K-quarter so every A element is fetched exactly once from HBM.
// B tiles are explicitly double-buffered in registers so their L2 latency
// overlaps the previous step's WMMAs. 3-way LDS reduction + relu epilogue.
// ---------------------------------------------------------------------------
template<int NT, bool RELU>
__global__ __launch_bounds__(512)
void spmm_wmma(const float* __restrict__ A, const __bf16* __restrict__ Y,
               float* __restrict__ C) {
  constexpr int F = NT * 16;
  __shared__ float red[3 * 4 * NT * 8 * 32];   // 48 KB (NT=4) / 24 KB (NT=2)

  const int lane  = threadIdx.x & 31;
  const int wave  = threadIdx.x >> 5;      // 0..15
  const int strip = wave & 3;
  const int kq    = wave >> 2;             // K quarter 0..3
  const int row16 = blockIdx.x * 64 + strip * 16;
  const int myrow = row16 + (lane & 15);
  const int hi8   = (lane >> 4) << 3;      // lanes>=16 take K-subgroup +8

  const float*  __restrict__ Arow  = A + (size_t)myrow * N_NODES;
  const __bf16* __restrict__ Ylane = Y + (size_t)lane * 16;
  const int kstart = kq * (N_NODES / 4);
  const int kend   = kstart + (N_NODES / 4);

  v8f acc[NT];
#pragma unroll
  for (int t = 0; t < NT; ++t) acc[t] = v8f{0.f,0.f,0.f,0.f,0.f,0.f,0.f,0.f};

  auto loadB = [&](int k0, v16bf* dst) {
    const __bf16* yb = Ylane + ((size_t)(k0 >> 5) * NT) * (32 * 16);
#pragma unroll
    for (int t = 0; t < NT; ++t)
      dst[t] = *(const v16bf*)(yb + (size_t)t * (32 * 16));
  };

  auto stepA = [&](int k0, const v16bf* b) {
    __builtin_prefetch(Arow + k0 + 128 + hi8, 0, 3);   // speculative -> GL2
    // A-operand 16x32 bf16 tile: per-lane K = k0+hi8+{0..7}, k0+16+hi8+{0..7}
    f4 f0 = *(const f4*)(Arow + k0 + hi8);
    f4 f1 = *(const f4*)(Arow + k0 + hi8 + 4);
    f4 f2 = *(const f4*)(Arow + k0 + hi8 + 16);
    f4 f3 = *(const f4*)(Arow + k0 + hi8 + 20);
    v16bf a;
#pragma unroll
    for (int i = 0; i < 4; ++i) {
      a[i]      = (__bf16)f0[i];
      a[4 + i]  = (__bf16)f1[i];
      a[8 + i]  = (__bf16)f2[i];
      a[12 + i] = (__bf16)f3[i];
    }
#pragma unroll
    for (int t = 0; t < NT; ++t)
      acc[t] = __builtin_amdgcn_wmma_f32_16x16x32_bf16(
          false, a, false, b[t], (short)0, acc[t], false, false);
  };

  // Software pipeline: B for step i+1 is loaded (into the other buffer)
  // before the WMMAs of step i consume the current buffer.
  v16bf ba[NT], bb[NT];
  loadB(kstart, ba);
  for (int k0 = kstart; k0 < kend; k0 += 64) {
    loadB(k0 + 32, bb);
    stepA(k0, ba);
    const int knext = (k0 + 64 < kend) ? (k0 + 64) : k0;  // clamped lookahead
    loadB(knext, ba);
    stepA(k0 + 32, bb);
  }

  // Cross-wave K reduction through LDS, then relu + store (C/D layout:
  // VGPR r -> row r + 8*(lane>=16), col = lane&15).
  if (kq != 0) {
#pragma unroll
    for (int t = 0; t < NT; ++t)
#pragma unroll
      for (int r = 0; r < 8; ++r)
        red[((((kq - 1) * 4 + strip) * NT + t) * 8 + r) * 32 + lane] =
            acc[t][r];
  }
  __syncthreads();
  if (kq == 0) {
#pragma unroll
    for (int t = 0; t < NT; ++t) {
#pragma unroll
      for (int r = 0; r < 8; ++r) {
        float v = acc[t][r];
#pragma unroll
        for (int s = 0; s < 3; ++s)
          v += red[(((s * 4 + strip) * NT + t) * 8 + r) * 32 + lane];
        if (RELU) v = fmaxf(v, 0.f);
        const int m = r + ((lane >> 4) << 3);
        C[(size_t)(row16 + m) * F + t * 16 + (lane & 15)] = v;
      }
    }
  }
}

// ---------------------------------------------------------------------------
extern "C" void kernel_launch(void* const* d_in, const int* in_sizes, int n_in,
                              void* d_out, int out_size, void* d_ws, size_t ws_size,
                              hipStream_t stream) {
  const float* A  = (const float*)d_in[0];   // [N,N]
  const float* X  = (const float*)d_in[1];   // [N,64]
  const float* W1 = (const float*)d_in[2];   // [64,64]
  const float* W2 = (const float*)d_in[3];   // [32,64]
  const float* W3 = (const float*)d_in[4];   // [32,32]
  float* out = (float*)d_out;                // [N,32]

  char* ws = (char*)d_ws;
  float*  h1 = (float*)(ws);                                  // 4 MB
  float*  h2 = (float*)(ws + (size_t)N_NODES * 64 * 4);       // 2 MB
  __bf16* Yp = (__bf16*)(ws + (size_t)N_NODES * 64 * 4
                            + (size_t)N_NODES * 32 * 4);      // 2 MB (reused)

  const dim3 blkPack(256);
  const dim3 blkGemm(512);
  const dim3 gPack64(N_NODES * 64 / 256);
  const dim3 gPack32(N_NODES * 32 / 256);
  const dim3 gGemm(N_NODES / 64);

  // Layer 1: h1 = relu(A @ (X @ W1^T))
  pack_feat<64, 64><<<gPack64, blkPack, 0, stream>>>(X, W1, Yp);
  spmm_wmma<4, true><<<gGemm, blkGemm, 0, stream>>>(A, Yp, h1);

  // Layer 2: h2 = relu(A @ (h1 @ W2^T))
  pack_feat<64, 32><<<gPack32, blkPack, 0, stream>>>(h1, W2, Yp);
  spmm_wmma<2, true><<<gGemm, blkGemm, 0, stream>>>(A, Yp, h2);

  // Layer 3: out = A @ (h2 @ W3^T)
  pack_feat<32, 32><<<gPack32, blkPack, 0, stream>>>(h2, W3, Yp);
  spmm_wmma<2, false><<<gGemm, blkGemm, 0, stream>>>(A, Yp, out);
}
